// mLSTM_cell_87883620811267
// MI455X (gfx1250) — matrix operand order repaired and checked by reading, not compiled, not run on hardware
//
#include <hip/hip_runtime.h>
#include <hip/hip_bf16.h>
#include <math.h>

#define DIM      1024
#define HEADS    16
#define HD       64
#define BS       4
#define SEQ      1024
#define GN_EPS   1e-5f
#define TAU_INV  (1.0f / 32.0f)   // 1/sqrt(1024)

typedef __attribute__((ext_vector_type(16))) __bf16 v16bf;
typedef __attribute__((ext_vector_type(8)))  float  v8f;

union BF16x16 { v16bf v; __bf16 b[16]; unsigned short u[16]; uint4 q[2]; };
union F32x8   { v8f   v; float e[8]; };

// f32 -> bf16, round-half-up (one add; ties differ from RNE by <=0.5 ulp bf16)
__device__ __forceinline__ unsigned short f2bf(float x) {
    return (unsigned short)((__float_as_uint(x) + 0x8000u) >> 16);
}

// Packed f32x2 -> bf16x2: bias both, then one V_PERM_B32 grabs the high halves.
__device__ __forceinline__ unsigned int f2bf_pk(float lo, float hi) {
    unsigned int ul = __float_as_uint(lo) + 0x8000u;
    unsigned int uh = __float_as_uint(hi) + 0x8000u;
#if defined(__has_builtin)
#if __has_builtin(__builtin_amdgcn_perm)
    return __builtin_amdgcn_perm(uh, ul, 0x07060302u);  // {uh[3],uh[2],ul[3],ul[2]}
#else
    return (ul >> 16) | (uh & 0xFFFF0000u);
#endif
#else
    return (ul >> 16) | (uh & 0xFFFF0000u);
#endif
}

__device__ __forceinline__ uint2 f2bf_x4(float4 x) {
    uint2 r;
    r.x = f2bf_pk(x.x, x.y);
    r.y = f2bf_pk(x.z, x.w);
    return r;
}

// ---------------------------------------------------------------------------
// Async global->LDS copy (CDNA5 GLOBAL_LOAD_ASYNC_TO_LDS_B128).
// Builtin params are vector-of-4-int pointers in addrspace(1)/(3); casts go
// through integers to avoid addrspace-cast restrictions.
// ---------------------------------------------------------------------------
#if defined(__has_builtin)
#if __has_builtin(__builtin_amdgcn_global_load_async_to_lds_b128) && \
    __has_builtin(__builtin_amdgcn_s_wait_asynccnt)
#define HAVE_ASYNC_CP 1
#endif
#endif
#ifndef HAVE_ASYNC_CP
#define HAVE_ASYNC_CP 0
#endif

#if HAVE_ASYNC_CP
typedef int v4i_ __attribute__((vector_size(16)));
typedef __attribute__((address_space(1))) v4i_ g_v4i;
typedef __attribute__((address_space(3))) v4i_ l_v4i;

__device__ __forceinline__ void async_cp16(void* lds_dst, const void* gsrc) {
    __builtin_amdgcn_global_load_async_to_lds_b128(
        (g_v4i*)(unsigned long long)gsrc,
        (l_v4i*)(unsigned int)(unsigned long long)lds_dst,
        0, 0);
}
__device__ __forceinline__ void async_wait() {
    __builtin_amdgcn_s_wait_asynccnt(0);
}
#else
__device__ __forceinline__ void async_cp16(void* lds_dst, const void* gsrc) {
    *(uint4*)lds_dst = *(const uint4*)gsrc;
}
__device__ __forceinline__ void async_wait() {}
#endif

// ---- WMMA fragment loaders (CDNA5 16x16x32 bf16, wave32) ------------------
// A (16x32): lane-half selects K base {0,8}; elements 0..7 -> K=base+e,
// elements 8..15 -> K=base+16+e.  Two contiguous 16B chunks per lane.
__device__ __forceinline__ v16bf load_a_frag(const unsigned short* lds,
                                             int row0, int k0, int ld, int lane) {
    BF16x16 f;
    const unsigned short* p =
        lds + (row0 + (lane & 15)) * ld + k0 + ((lane >> 4) << 3);
    f.q[0] = *(const uint4*)(p);
    f.q[1] = *(const uint4*)(p + 16);
    return f.v;
}

// B (32x16) from a K-major tile Bsw[n][k]: lane n = lane&15, K base =
// k0 + 8*half; elements 0..7 -> K=base+e, 8..15 -> K=base+16+e.  Two 16B chunks per lane.
__device__ __forceinline__ v16bf load_b_frag_kmaj(const unsigned short* lds,
                                                  int n0, int k0, int ld, int lane) {
    BF16x16 f;
    const unsigned short* p =
        lds + (n0 + (lane & 15)) * ld + k0 + ((lane >> 4) << 3);
    f.q[0] = *(const uint4*)(p);
    f.q[1] = *(const uint4*)(p + 16);
    return f.v;
}

__device__ __forceinline__ v8f wmma_bf16(v16bf a, v16bf b, v8f c) {
    return __builtin_amdgcn_wmma_f32_16x16x32_bf16(false, a, false, b,
                                                   (short)0, c, false, false);
}

// ===========================================================================
// Kernel 1: gate pre-activations + log-sigmoid cumsum
// ===========================================================================
__global__ __launch_bounds__(256) void gates_kernel(
    const float* __restrict__ q, const float* __restrict__ k,
    const float* __restrict__ Wi_w, const float* __restrict__ Wi_b,
    const float* __restrict__ Wf_w, const float* __restrict__ Wf_b,
    float* __restrict__ cf, float* __restrict__ itil)
{
    __shared__ float ls_f[SEQ];
    __shared__ float it_s[SEQ];
    const int bh = blockIdx.x;
    const int b = bh / HEADS, h = bh % HEADS;
    const int lane = threadIdx.x & 31, wave = threadIdx.x >> 5;

    for (int s = wave; s < SEQ; s += 8) {
        const float4* qr = (const float4*)(q + ((size_t)b * SEQ + s) * DIM);
        const float4* kr = (const float4*)(k + ((size_t)b * SEQ + s) * DIM);
        const float4* wi = (const float4*)(Wi_w + (size_t)h * DIM);
        const float4* wf = (const float4*)(Wf_w + (size_t)h * DIM);
        float ai = 0.f, af = 0.f;
        for (int d = lane; d < DIM / 4; d += 32) {
            float4 xq = qr[d], xk = kr[d], yi = wi[d], yf = wf[d];
            ai = fmaf(xq.x, yi.x, fmaf(xq.y, yi.y, fmaf(xq.z, yi.z, fmaf(xq.w, yi.w, ai))));
            af = fmaf(xk.x, yf.x, fmaf(xk.y, yf.y, fmaf(xk.z, yf.z, fmaf(xk.w, yf.w, af))));
        }
#pragma unroll
        for (int off = 16; off; off >>= 1) {
            ai += __shfl_xor(ai, off, 32);
            af += __shfl_xor(af, off, 32);
        }
        if (lane == 0) {
            float ft = af + Wf_b[h];
            ls_f[s] = fminf(ft, 0.f) - log1pf(__expf(-fabsf(ft)));  // logsigmoid
            it_s[s] = ai + Wi_b[h];
        }
    }
    __syncthreads();
    if (threadIdx.x == 0) {
        float run = 0.f;
        float* cfo = cf + (size_t)bh * SEQ;
        for (int s = 0; s < SEQ; ++s) { run += ls_f[s]; cfo[s] = run; }
    }
    for (int s = threadIdx.x; s < SEQ; s += 256)
        itil[(size_t)bh * SEQ + s] = it_s[s];
}

// ===========================================================================
// Kernel 2: projection GEMM  Y = X · Wᵀ + bias (bf16 WMMA, f32 accum)
//   Q/K written head-major [b][h][s][hd]; V written transposed [b][h][hd][s]
//   Block tile 64x64, K-step 32, 4 waves (2x2), each wave 32x32.
// ===========================================================================
#define PBM 64
#define PBN 64
#define PBK 32

__global__ __launch_bounds__(128) void proj_kernel(
    const float* __restrict__ X, const float* __restrict__ W,
    const float* __restrict__ bias, unsigned short* __restrict__ Y,
    const int transposeOut)
{
    __shared__ __align__(16) unsigned short As[PBM * PBK];   // A[m][k]
    __shared__ __align__(16) unsigned short Bs[PBN * PBK];   // Bsw[n][k] = W natural

    const int tid = threadIdx.x;
    const int lane = tid & 31, wave = tid >> 5;
    const int nb = DIM / PBN;                 // 16
    const int m0 = ((int)blockIdx.x / nb) * PBM;
    const int n0 = ((int)blockIdx.x % nb) * PBN;
    const int wm = (wave >> 1) * 32;
    const int wn = (wave & 1) * 32;

    F32x8 acc[2][2];
#pragma unroll
    for (int tm = 0; tm < 2; ++tm)
#pragma unroll
        for (int tn = 0; tn < 2; ++tn)
#pragma unroll
            for (int e = 0; e < 8; ++e) acc[tm][tn].e[e] = 0.f;

    for (int k0 = 0; k0 < DIM; k0 += PBK) {
        // 64 rows x 32 cols, float4 loads + packed bf16x4 stores
#pragma unroll
        for (int e = tid; e < PBM * PBK / 4; e += 128) {
            int r = e >> 3, c = (e & 7) << 2;
            float4 xa = *(const float4*)(X + (size_t)(m0 + r) * DIM + k0 + c);
            float4 xb = *(const float4*)(W + (size_t)(n0 + r) * DIM + k0 + c);
            *(uint2*)&As[r * PBK + c] = f2bf_x4(xa);
            *(uint2*)&Bs[r * PBK + c] = f2bf_x4(xb);
        }
        __syncthreads();

        v16bf a0 = load_a_frag(As, wm,      0, PBK, lane);
        v16bf a1 = load_a_frag(As, wm + 16, 0, PBK, lane);
        v16bf b0 = load_b_frag_kmaj(Bs, wn,      0, PBK, lane);
        v16bf b1 = load_b_frag_kmaj(Bs, wn + 16, 0, PBK, lane);

        acc[0][0].v = wmma_bf16(a0, b0, acc[0][0].v);
        acc[0][1].v = wmma_bf16(a0, b1, acc[0][1].v);
        acc[1][0].v = wmma_bf16(a1, b0, acc[1][0].v);
        acc[1][1].v = wmma_bf16(a1, b1, acc[1][1].v);
        __syncthreads();
    }

    const int half = lane >> 4, jc = lane & 15;
#pragma unroll
    for (int tm = 0; tm < 2; ++tm)
#pragma unroll
        for (int tn = 0; tn < 2; ++tn)
#pragma unroll
            for (int e = 0; e < 8; ++e) {
                int m = m0 + wm + tm * 16 + e + (half << 3);
                int n = n0 + wn + tn * 16 + jc;
                float y = acc[tm][tn].e[e] + bias[n];
                int b = m >> 10, s = m & 1023, h = n >> 6, hd = n & 63;
                size_t dst = transposeOut
                    ? ((size_t)(b * HEADS + h) * HD + hd) * SEQ + s
                    : ((size_t)(b * HEADS + h) * SEQ + s) * HD + hd;
                Y[dst] = f2bf(y);
            }
}

// ===========================================================================
// Kernel 3: flash-style mLSTM attention + normalizer + fused GroupNorm
//   grid (BS*HEADS, SEQ/64), 128 threads (4 waves, each owns 16 query rows)
//   Ks: K tile natural [j][d]  (B-frag for S, K-major per lane)
//   Vs: V tile [d][j]          (B-frag for H, fed by pre-transposed V)
// ===========================================================================
__global__ __launch_bounds__(128) void mlstm_attn_kernel(
    const unsigned short* __restrict__ Qh, const unsigned short* __restrict__ Kh,
    const unsigned short* __restrict__ Vt,
    const float* __restrict__ cf_g, const float* __restrict__ it_g,
    const float* __restrict__ gamma, const float* __restrict__ beta,
    float* __restrict__ out)
{
    __shared__ __align__(16) unsigned short Ks[64 * HD];      // [j][d] (also Q staging)
    __shared__ __align__(16) unsigned short Vs[HD * 64];      // [d][j]
    __shared__ __align__(16) unsigned short Ps[4 * 16 * 64];  // per-wave P strips
    __shared__ float cfi[64], cfj[64], itj[64];

    const int bh = blockIdx.x;
    const int b = bh / HEADS, h = bh % HEADS;
    const int i0 = (int)blockIdx.y * 64;
    const int tid = threadIdx.x, lane = tid & 31, wave = tid >> 5;
    const int half = lane >> 4, jc = lane & 15;

    const unsigned short* Qb = Qh + (size_t)bh * SEQ * HD;
    const unsigned short* Kb = Kh + (size_t)bh * SEQ * HD;
    const unsigned short* Vb = Vt + (size_t)bh * HD * SEQ;    // transposed: [hd][s]
    const float* cfb = cf_g + (size_t)bh * SEQ;
    const float* itb = it_g + (size_t)bh * SEQ;

    // ---- stage Q tile (contiguous 8KB) into Ks, build A fragments ---------
    {
        const unsigned short* qsrc = Qb + (size_t)i0 * HD;
#pragma unroll
        for (int c = tid; c < 512; c += 128)
            async_cp16(&Ks[c * 8], qsrc + c * 8);
    }
    if (tid < 64) cfi[tid] = cfb[i0 + tid];
    async_wait();
    __syncthreads();
    v16bf qf0 = load_a_frag(Ks, wave * 16,  0, HD, lane);
    v16bf qf1 = load_a_frag(Ks, wave * 16, 32, HD, lane);
    __syncthreads();

    float m_r[8], rs[8];
    F32x8 accH[4];
#pragma unroll
    for (int e = 0; e < 8; ++e) { m_r[e] = -1e30f; rs[e] = 0.f; }
#pragma unroll
    for (int nt = 0; nt < 4; ++nt)
#pragma unroll
        for (int e = 0; e < 8; ++e) accH[nt].e[e] = 0.f;

    for (int j0 = 0; j0 <= i0; j0 += 64) {
        // ---- async-load K tile (contiguous) and V tile (64 strided rows) --
        {
            const unsigned short* ksrc = Kb + (size_t)j0 * HD;
#pragma unroll
            for (int c = tid; c < 512; c += 128)
                async_cp16(&Ks[c * 8], ksrc + c * 8);
#pragma unroll
            for (int c = tid; c < 512; c += 128) {
                int d = c >> 3, part = c & 7;
                async_cp16(&Vs[d * 64 + part * 8],
                           Vb + (size_t)d * SEQ + j0 + part * 8);
            }
        }
        if (tid < 64) { cfj[tid] = cfb[j0 + tid]; itj[tid] = itb[j0 + tid]; }
        // prefetch next K tile (one 128B line per lane over first 2 waves)
        if (j0 + 64 <= i0 && tid < 64)
            __builtin_prefetch((const char*)(Kb + (size_t)(j0 + 64) * HD) + tid * 128, 0, 0);
        async_wait();
        __syncthreads();

        // ---- S = Q·Kᵀ and decay logits Dtilde -----------------------------
        float s_t[4][8], dt[4][8];
#pragma unroll
        for (int jt = 0; jt < 4; ++jt) {
            v16bf kb0 = load_b_frag_kmaj(Ks, jt * 16,  0, HD, lane);
            v16bf kb1 = load_b_frag_kmaj(Ks, jt * 16, 32, HD, lane);
            F32x8 c;
#pragma unroll
            for (int e = 0; e < 8; ++e) c.e[e] = 0.f;
            c.v = wmma_bf16(qf0, kb0, c.v);
            c.v = wmma_bf16(qf1, kb1, c.v);

            const int j = j0 + jt * 16 + jc;
            const float cfjv = cfj[jt * 16 + jc];
            const float itjv = itj[jt * 16 + jc];
#pragma unroll
            for (int e = 0; e < 8; ++e) {
                const int r = wave * 16 + e + (half << 3);
                dt[jt][e]  = (j <= i0 + r) ? (cfi[r] - cfjv + itjv) : -1e30f;
                s_t[jt][e] = c.e[e];
            }
        }

        // ---- online row max + rescale -------------------------------------
        float sc[8];
#pragma unroll
        for (int e = 0; e < 8; ++e) {
            float mx = fmaxf(fmaxf(dt[0][e], dt[1][e]), fmaxf(dt[2][e], dt[3][e]));
#pragma unroll
            for (int off = 8; off; off >>= 1) mx = fmaxf(mx, __shfl_xor(mx, off, 32));
            float mn = fmaxf(m_r[e], mx);
            sc[e] = __expf(m_r[e] - mn);
            m_r[e] = mn;
            rs[e] *= sc[e];
        }
#pragma unroll
        for (int nt = 0; nt < 4; ++nt)
#pragma unroll
            for (int e = 0; e < 8; ++e) accH[nt].e[e] *= sc[e];

        // ---- P = S * exp(Dtilde - m); rowsum; re-layout via LDS -----------
        float pl[4][8];
#pragma unroll
        for (int jt = 0; jt < 4; ++jt)
#pragma unroll
            for (int e = 0; e < 8; ++e) {
                float p = s_t[jt][e] * __expf(dt[jt][e] - m_r[e]);
                pl[jt][e] = p;
                Ps[(wave * 16 + e + (half << 3)) * 64 + jt * 16 + jc] = f2bf(p);
            }
#pragma unroll
        for (int e = 0; e < 8; ++e) {
            float su = (pl[0][e] + pl[1][e]) + (pl[2][e] + pl[3][e]);
#pragma unroll
            for (int off = 8; off; off >>= 1) su += __shfl_xor(su, off, 32);
            rs[e] += su;
        }
        __syncthreads();

        // ---- H += P · V ---------------------------------------------------
        v16bf pa0 = load_a_frag(Ps + wave * 16 * 64, 0,  0, 64, lane);
        v16bf pa1 = load_a_frag(Ps + wave * 16 * 64, 0, 32, 64, lane);
#pragma unroll
        for (int nt = 0; nt < 4; ++nt) {
            v16bf vb0 = load_b_frag_kmaj(Vs, nt * 16,  0, 64, lane);
            v16bf vb1 = load_b_frag_kmaj(Vs, nt * 16, 32, 64, lane);
            accH[nt].v = wmma_bf16(pa0, vb0, accH[nt].v);
            accH[nt].v = wmma_bf16(pa1, vb1, accH[nt].v);
        }
        __syncthreads();
    }

    // ---- mLSTM normalizer + fused GroupNorm + store -----------------------
    float dinv[8];
#pragma unroll
    for (int e = 0; e < 8; ++e) {
        float rsf = rs[e] * TAU_INV;
        float denom = fmaxf(fabsf(rsf), __expf(-m_r[e])) + 1e-8f;
        dinv[e] = TAU_INV / denom;
    }
    float hv[4][8];
#pragma unroll
    for (int nt = 0; nt < 4; ++nt)
#pragma unroll
        for (int e = 0; e < 8; ++e) hv[nt][e] = accH[nt].e[e] * dinv[e];

    float mean_[8], rstd_[8];
#pragma unroll
    for (int e = 0; e < 8; ++e) {
        float s1 = (hv[0][e] + hv[1][e]) + (hv[2][e] + hv[3][e]);
        float s2 = hv[0][e]*hv[0][e] + hv[1][e]*hv[1][e]
                 + hv[2][e]*hv[2][e] + hv[3][e]*hv[3][e];
#pragma unroll
        for (int off = 8; off; off >>= 1) {
            s1 += __shfl_xor(s1, off, 32);
            s2 += __shfl_xor(s2, off, 32);
        }
        float mean = s1 * (1.f / 64.f);
        float var  = s2 * (1.f / 64.f) - mean * mean;
        mean_[e] = mean;
        rstd_[e] = rsqrtf(var + GN_EPS);
    }
#pragma unroll
    for (int nt = 0; nt < 4; ++nt)
#pragma unroll
        for (int e = 0; e < 8; ++e) {
            int i  = i0 + wave * 16 + e + (half << 3);
            int d  = nt * 16 + jc;
            int ch = h * HD + d;
            float y = (hv[nt][e] - mean_[e]) * rstd_[e] * gamma[ch] + beta[ch];
            out[((size_t)b * SEQ + i) * DIM + ch] = y;
        }
}

// ===========================================================================
extern "C" void kernel_launch(void* const* d_in, const int* in_sizes, int n_in,
                              void* d_out, int out_size, void* d_ws, size_t ws_size,
                              hipStream_t stream) {
    const float* q    = (const float*)d_in[0];
    const float* k    = (const float*)d_in[1];
    const float* v    = (const float*)d_in[2];
    const float* Wq_w = (const float*)d_in[3];
    const float* Wq_b = (const float*)d_in[4];
    const float* Wk_w = (const float*)d_in[5];
    const float* Wk_b = (const float*)d_in[6];
    const float* Wv_w = (const float*)d_in[7];
    const float* Wv_b = (const float*)d_in[8];
    const float* Wi_w = (const float*)d_in[9];
    const float* Wi_b = (const float*)d_in[10];
    const float* Wf_w = (const float*)d_in[11];
    const float* Wf_b = (const float*)d_in[12];
    const float* gam  = (const float*)d_in[13];
    const float* bet  = (const float*)d_in[14];
    float* out = (float*)d_out;

    const size_t nQKV = (size_t)BS * HEADS * SEQ * HD;      // 4M elems
    unsigned short* Qh = (unsigned short*)d_ws;
    unsigned short* Kh = Qh + nQKV;
    unsigned short* Vh = Kh + nQKV;                          // transposed layout
    float* cf   = (float*)(Vh + nQKV);
    float* itil = cf + (size_t)BS * HEADS * SEQ;

    gates_kernel<<<BS * HEADS, 256, 0, stream>>>(q, k, Wi_w, Wi_b, Wf_w, Wf_b,
                                                 cf, itil);

    const int projBlocks = (BS * SEQ / PBM) * (DIM / PBN);  // 1024
    proj_kernel<<<projBlocks, 128, 0, stream>>>(q, Wq_w, Wq_b, Qh, 0);
    proj_kernel<<<projBlocks, 128, 0, stream>>>(k, Wk_w, Wk_b, Kh, 0);
    proj_kernel<<<projBlocks, 128, 0, stream>>>(v, Wv_w, Wv_b, Vh, 1);

    dim3 g(BS * HEADS, SEQ / 64);
    mlstm_attn_kernel<<<g, 128, 0, stream>>>(Qh, Kh, Vh, cf, itil, gam, bet, out);
}
